// SL_CSC_FISTA_30855045054849
// MI455X (gfx1250) — matrix operand order repaired and checked by reading, not compiled, not run on hardware
//
#include <hip/hip_runtime.h>
#include <cmath>

// ---------------------------------------------------------------------------
// FISTA convolutional sparse coding for MI455X (gfx1250, wave32).
//
// grad_step:  Z <- Z - step * Dt( D(Z) - Y )
//
// conv_D : windowed-Toeplitz implicit GEMM.
//   Output tile 16x16 (M = output rows u, N = output cols v).
//   K = (atom a, tap-row di, 25-col absolute window padded to 32)
//       -> ONE v_wmma_f32_16x16x32_bf16 per (a,di): 1000 wmma / tile.
//   A fragment = 2x ds_load_b128 from LDS staged as bf16 column-pairs
//   (all v-tiles share odd pair parity -> groups of 4 dwords, 16B aligned).
//   B fragment = per-(a,di) 32x16 Toeplitz band of flipped W (1MB table,
//   identical for all 8 waves of a WG -> WGP$-hot global_load_b128 x2).
//
// conv_Dt: implicit GEMM, M = 16 output cols, N = atom tiles (7x16),
//   K = 100 taps padded to 128 (4 wmma), fused in-place Z update.
// ---------------------------------------------------------------------------

typedef __attribute__((ext_vector_type(16))) __bf16 v16bf;
typedef __attribute__((ext_vector_type(8)))  float  v8f;
typedef __attribute__((ext_vector_type(8)))  int    v8i;
typedef __attribute__((ext_vector_type(4)))  int    v4i;

#define STEPSZ 0.1f
#define NXELEM (16UL * 100UL * 119UL * 119UL)   // 22,657,600
#define CHD 10                                  // atoms per LDS chunk (conv_D)

__device__ __forceinline__ unsigned pk_bf16(float x, float y) {
  unsigned a = __builtin_bit_cast(unsigned, x);
  unsigned b = __builtin_bit_cast(unsigned, y);
  a = (a + 0x7FFFu + ((a >> 16) & 1u)) >> 16;   // RNE f32 -> bf16
  b = (b + 0x7FFFu + ((b >> 16) & 1u)) >> 16;
  return (b << 16) | (a & 0xFFFFu);
}

// WMMA bf16 per-lane layouts (CDNA5 ISA 7.12.2):
//  A 16x32: lane<16 M=lane, dword d: d<4 -> K=h*8+2d ; d>=4 -> K=16+h*8+2(d-4)   (h=lane>>4)
//  B 32x16: N=lane&15, dword v: v<4 -> K=h*8+2v ; v>=4 -> K=16+h*8+2(v-4)  (same as A)
//  C 16x16: lane L: N=L&15, VGPR r: M=r+8*(L>>4)

// --- weight table prep -------------------------------------------------------
// tabD2[a][di][lane][v] : conv_D Toeplitz B frags (flipped kernel), 256,000 dw
// tabDt[nt][s][lane][v] : conv_Dt B frags, Wt[a,0,p,q] = W[a,0,q,p],   7,168 dw
__global__ void prep_tables(const float* __restrict__ W,
                            unsigned* __restrict__ tabD2,
                            unsigned* __restrict__ tabDt) {
  const int stride = blockDim.x * gridDim.x;
  const int t0 = blockIdx.x * blockDim.x + threadIdx.x;
  for (int idx = t0; idx < 100 * 10 * 32 * 8; idx += stride) {
    int v = idx & 7, lane = (idx >> 3) & 31;
    int rest = idx >> 8;
    int di = rest % 10, a = rest / 10;
    int nv = lane & 15, hh = lane >> 4;
    int K0 = (v < 4) ? (hh * 8 + 2 * v) : (16 + hh * 8 + 2 * (v - 4));   // B-fragment K ordering
    float w0 = 0.f, w1 = 0.f;
    int d0 = K0 - nv, d1 = K0 + 1 - nv;           // dj = kk - nv
    if (d0 >= 0 && d0 <= 9) w0 = W[a * 100 + (9 - di) * 10 + (9 - d0)];
    if (d1 >= 0 && d1 <= 9) w1 = W[a * 100 + (9 - di) * 10 + (9 - d1)];
    tabD2[idx] = pk_bf16(w0, w1);
  }
  for (int idx = t0; idx < 7 * 4 * 32 * 8; idx += stride) {
    int v = idx & 7, lane = (idx >> 3) & 31, s = (idx >> 8) & 3, nt = idx >> 10;
    int a = nt * 16 + (lane & 15), hh = lane >> 4;
    int K0 = s * 32 + ((v < 4) ? (hh * 8 + 2 * v) : (16 + hh * 8 + 2 * (v - 4)));   // k = dj*10 + di
    float w0 = 0.f, w1 = 0.f;
    if (a < 100) {
      if (K0 < 100)     { int di = K0 % 10,     dj = K0 / 10;     w0 = W[a*100 + dj*10 + di]; }
      if (K0 + 1 < 100) { int di = (K0+1) % 10, dj = (K0+1) / 10; w1 = W[a*100 + dj*10 + di]; }
    }
    tabDt[idx] = pk_bf16(w0, w1);
  }
}

// --- conv_D fused residual: R[n,u,v] = (D Z)[n,u,v] - Y[n,u,v] --------------
// grid (8,16): (u-tile, n); block 256 = 8 waves; wave w -> v-tile [16w,16w+16)
// LDS: CHD atoms x 25 rows x 72 col-pairs (bf16x2), abs col b = 2*ci - 9.
__global__ __launch_bounds__(256) void conv_d_kernel(
    const float* __restrict__ Z, const float* __restrict__ Y,
    const unsigned* __restrict__ tabD2, float* __restrict__ R) {
  __shared__ __align__(16) unsigned ldsz[CHD * 25 * 72];   // 72,000 B
  const int u0 = blockIdx.x * 16;
  const int n  = blockIdx.y;
  const int tid = threadIdx.x;
  const int lane = tid & 31;
  const int wv = tid >> 5;
  const int m = lane & 15;
  const int h = lane >> 4;
  const int ci0 = 8 * wv + 4 * h;                 // group-of-4 aligned (mod 4 == 0)
  v8f C = {};
  for (int chunk = 0; chunk < 10; ++chunk) {
    const int a0 = chunk * CHD;
    __syncthreads();
    for (int idx = tid; idx < CHD * 25 * 72; idx += 256) {
      int ci = idx % 72;
      int rest = idx / 72;
      int rr = rest % 25;
      int al = rest / 25;
      int row = u0 - 9 + rr;
      int b = 2 * ci - 9;
      const float* zp = Z + (size_t)(n * 100 + a0 + al) * (119 * 119);
      bool rok = (row >= 0) & (row < 119);
      float z0 = (rok && b >= 0 && b < 119)         ? zp[row * 119 + b]     : 0.f;
      float z1 = (rok && b + 1 >= 0 && b + 1 < 119) ? zp[row * 119 + b + 1] : 0.f;
      ldsz[idx] = pk_bf16(z0, z1);
    }
    __syncthreads();
    for (int al = 0; al < CHD; ++al) {
      const int a = a0 + al;
      const unsigned* bt = tabD2 + ((size_t)a * 10 * 32 + lane) * 8;
#pragma unroll
      for (int di = 0; di < 10; ++di) {
        const unsigned* ap = &ldsz[(al * 25 + m + di) * 72];
        v4i x0 = *(const v4i*)(ap + ci0);          // ds_load_b128
        v4i x1 = *(const v4i*)(ap + ci0 + 8);      // ds_load_b128
        v8i av = __builtin_shufflevector(x0, x1, 0, 1, 2, 3, 4, 5, 6, 7);
        v8i bv = *(const v8i*)(bt + (size_t)di * 256);
        C = __builtin_amdgcn_wmma_f32_16x16x32_bf16(
                false, __builtin_bit_cast(v16bf, av),
                false, __builtin_bit_cast(v16bf, bv),
                (short)0, C, false, false);
      }
    }
  }
  // C: lane holds column v = 16*wv + m, rows u = u0 + 8h + r
  const size_t base = (size_t)n * 16384 + (size_t)(u0 + 8 * h) * 128 + 16 * wv + m;
#pragma unroll
  for (int r = 0; r < 8; ++r) {
    const size_t idx = base + (size_t)r * 128;
    R[idx] = C[r] - Y[idx];
  }
}

// --- conv_Dt fused in-place update: Z[n,a,i,j] -= step * (Dt R)[n,a,i,j] ----
// grid (119,16); block 256 = 8 waves; wave w -> j-tile [16w,16w+16)
__global__ __launch_bounds__(256) void conv_dt_kernel(
    const float* __restrict__ Rbuf, const unsigned* __restrict__ tabDt,
    float* __restrict__ Zio) {
  __shared__ unsigned ldsr[137 * 5];
  const int i = blockIdx.x;
  const int n = blockIdx.y;
  const int tid = threadIdx.x;
  const int lane = tid & 31;
  const int j0 = (tid >> 5) * 16;
  const int m = lane & 15;
  const int h = lane >> 4;
  for (int idx = tid; idx < 137 * 5; idx += 256) {
    int cidx = idx % 137;
    int t = idx / 137;
    const float* rp = Rbuf + (size_t)n * 16384;
    int r0 = i + 2 * t;
    float z0 = (cidx < 128 && r0 < 128)     ? rp[r0 * 128 + cidx]       : 0.f;
    float z1 = (cidx < 128 && r0 + 1 < 128) ? rp[(r0 + 1) * 128 + cidx] : 0.f;
    ldsr[cidx * 5 + t] = pk_bf16(z0, z1);
  }
  __syncthreads();
  for (int nt = 0; nt < 7; ++nt) {
    v8f C = {};
#pragma unroll
    for (int s = 0; s < 4; ++s) {
      v8i av;
#pragma unroll
      for (int d = 0; d < 8; ++d) {
        int K = s * 32 + ((d < 4) ? (h * 8 + 2 * d) : (16 + h * 8 + 2 * (d - 4)));
        unsigned val = 0u;
        if (K < 100) {
          int di = K % 10, dj = K / 10;
          val = ldsr[(j0 + m + dj) * 5 + (di >> 1)];
        }
        av[d] = (int)val;
      }
      const v8i bv = *(const v8i*)(tabDt + (((nt * 4 + s) * 32 + lane) << 3));
      C = __builtin_amdgcn_wmma_f32_16x16x32_bf16(
              false, __builtin_bit_cast(v16bf, av),
              false, __builtin_bit_cast(v16bf, bv),
              (short)0, C, false, false);
    }
    const int a = nt * 16 + m;
    if (a < 100) {
      const int jb = j0 + h * 8;
      float* zp = Zio + ((size_t)(n * 100 + a) * 119 + i) * 119;
#pragma unroll
      for (int r = 0; r < 8; ++r) {
        int j = jb + r;
        if (j < 119) zp[j] -= STEPSZ * C[r];
      }
    }
  }
}

// --- fused soft-threshold + FISTA momentum (float4 vectorized) --------------
__global__ void fista_elem(const float* __restrict__ ST, const float* __restrict__ X1,
                           float* __restrict__ X2, float* Zout, float c, int n4) {
  int idx = blockIdx.x * 256 + threadIdx.x;
  if (idx >= n4) return;
  float4 st = ((const float4*)ST)[idx];
  float4 x1 = ((const float4*)X1)[idx];
  float4 x2, z;
  float t;
  t = fabsf(st.x) - STEPSZ; x2.x = t > 0.f ? copysignf(t, st.x) : 0.f; z.x = x2.x + (x2.x - x1.x) * c;
  t = fabsf(st.y) - STEPSZ; x2.y = t > 0.f ? copysignf(t, st.y) : 0.f; z.y = x2.y + (x2.y - x1.y) * c;
  t = fabsf(st.z) - STEPSZ; x2.z = t > 0.f ? copysignf(t, st.z) : 0.f; z.z = x2.z + (x2.z - x1.z) * c;
  t = fabsf(st.w) - STEPSZ; x2.w = t > 0.f ? copysignf(t, st.w) : 0.f; z.w = x2.w + (x2.w - x1.w) * c;
  ((float4*)X2)[idx] = x2;
  ((float4*)Zout)[idx] = z;
}

// ---------------------------------------------------------------------------
extern "C" void kernel_launch(void* const* d_in, const int* in_sizes, int n_in,
                              void* d_out, int out_size, void* d_ws, size_t ws_size,
                              hipStream_t stream) {
  const float* Y  = (const float*)d_in[0];   // 16x1x128x128
  const float* X0 = (const float*)d_in[1];   // 16x100x119x119
  const float* W  = (const float*)d_in[2];   // 100x1x10x10
  float* out = (float*)d_out;

  float* Zbuf = (float*)d_ws;                 // 90.6 MB : Z / ST (in place)
  float* Xa   = Zbuf + NXELEM;                // 90.6 MB : ping-pong X2
  float* Rbuf = Xa + NXELEM;                  // 1 MB    : residual
  unsigned* tabD2 = (unsigned*)(Rbuf + 16 * 128 * 128);   // 1 MB Toeplitz B
  unsigned* tabDt = tabD2 + 100 * 10 * 32 * 8;            // 28 KB

  // FISTA momentum coefficients (data independent)
  float coef[15];
  double tcur = 1.0;
  for (int k = 0; k < 15; ++k) {
    double t1 = tcur;
    tcur = (1.0 + sqrt(1.0 + 4.0 * t1 * t1)) / 2.0;
    coef[k] = (float)((t1 - 1.0) / tcur);
  }

  const int n4 = (int)(NXELEM / 4);
  const int eg = (n4 + 255) / 256;

  prep_tables<<<256, 256, 0, stream>>>(W, tabD2, tabDt);
  hipMemcpyAsync(Zbuf, X0, NXELEM * sizeof(float), hipMemcpyDeviceToDevice, stream);

  // initial grad step: Zbuf <- grad_step(X0)
  conv_d_kernel <<<dim3(8, 16),   256, 0, stream>>>(Zbuf, Y, tabD2, Rbuf);
  conv_dt_kernel<<<dim3(119, 16), 256, 0, stream>>>(Rbuf, tabDt, Zbuf);

  const float* X1 = X0;
  for (int it = 0; it < 15; ++it) {
    float* X2 = (it % 2 == 0) ? out : Xa;    // it=14 (last) lands in d_out
    fista_elem<<<eg, 256, 0, stream>>>(Zbuf, X1, X2, Zbuf, coef[it], n4);
    if (it < 14) {                           // last grad step of reference is discarded
      conv_d_kernel <<<dim3(8, 16),   256, 0, stream>>>(Zbuf, Y, tabD2, Rbuf);
      conv_dt_kernel<<<dim3(119, 16), 256, 0, stream>>>(Rbuf, tabDt, Zbuf);
    }
    X1 = X2;
  }
}
